// SwinTransformerBlock3D_84782654423463
// MI455X (gfx1250) — compile-verified, matrix operand order checked by reading, not run on hardware
//
#include <hip/hip_runtime.h>
#include <math.h>

#define CC   384
#define C3   1152
#define HID  1536
#define NHD  12
#define HDD  32
#define DHW  50176
#define NTOK 100352L
#define KCH  384                 // K-chunk staged in LDS per block
#define KPAD 8                   // row pad (halfs) -> conflict-free ds_load_b128

typedef _Float16 half_t;
typedef _Float16 v16h __attribute__((ext_vector_type(16)));
typedef float    v8f  __attribute__((ext_vector_type(8)));
typedef float    f4   __attribute__((ext_vector_type(4)));

__device__ __forceinline__ float wred(float v) {
#pragma unroll
  for (int off = 16; off > 0; off >>= 1) v += __shfl_xor(v, off, 32);
  return v;
}

// Load one 16x32 f16 WMMA operand tile slice for this lane (two 16B chunks).
__device__ __forceinline__ v16h ldtile(const half_t* __restrict__ p) {
  v16h r;
  ((f4*)&r)[0] = *(const f4*)(p);
  ((f4*)&r)[1] = *(const f4*)(p + 16);
  return r;
}

// ---------- f32 -> f16 weight conversion ----------
__global__ void k_cvt(const float* __restrict__ in, half_t* __restrict__ out, int n) {
  int i = blockIdx.x * 256 + threadIdx.x;
  if (i < n) out[i] = (half_t)in[i];
}

// ---------- permute (B,C,D,H,W) -> (N,C) + LayerNorm1 ----------
__global__ void __launch_bounds__(256) k_ln1(const float* __restrict__ x,
                                             float* __restrict__ xt,
                                             half_t* __restrict__ y16,
                                             const float* __restrict__ g,
                                             const float* __restrict__ b) {
  __shared__ float tile[CC * 33];
  __shared__ float smu[32], srs[32];
  const int tid = threadIdx.x;
  const long n0 = (long)blockIdx.x * 32;
  const long bi = n0 / DHW;
  const long s0 = n0 - bi * DHW;
  const float* xb = x + bi * (long)CC * DHW + s0;
  for (int e = tid; e < CC * 32; e += 256) {        // coalesced over spatial
    int c = e >> 5, t = e & 31;
    tile[c * 33 + t] = xb[(long)c * DHW + t];
  }
  __syncthreads();
  const int wave = tid >> 5, lane = tid & 31;
#pragma unroll
  for (int rep = 0; rep < 4; ++rep) {
    int t = wave * 4 + rep;
    float s = 0.f, q = 0.f;
#pragma unroll
    for (int i = 0; i < 12; ++i) {
      float v = tile[(lane + 32 * i) * 33 + t];
      s += v; q += v * v;
    }
    s = wred(s); q = wred(q);
    if (lane == 0) {
      float mu = s * (1.0f / CC);
      float var = q * (1.0f / CC) - mu * mu;
      smu[t] = mu; srs[t] = rsqrtf(var + 1e-5f);
    }
  }
  __syncthreads();
  for (int e = tid; e < CC * 32; e += 256) {        // coalesced over channels
    int t = e / CC, c = e - t * CC;
    float v = tile[c * 33 + t];
    long idx = (n0 + t) * CC + c;
    xt[idx] = v;
    y16[idx] = (half_t)((v - smu[t]) * srs[t] * g[c] + b[c]);
  }
}

// ---------- per-token cross-head attention + gamma1 residual + LN2 ----------
__global__ void __launch_bounds__(128) k_attn(const half_t* __restrict__ qkv,
                                              float* __restrict__ xt,
                                              const float* __restrict__ gamma1,
                                              const float* __restrict__ g2,
                                              const float* __restrict__ b2,
                                              half_t* __restrict__ y2) {
  __shared__ float sQ[4][C3];
  __shared__ float sA[4][NHD * NHD];
  const int w = threadIdx.x >> 5, lane = threadIdx.x & 31;
  const long n = (long)blockIdx.x * 4 + w;
  const half_t* qr = qkv + n * C3;
#pragma unroll
  for (int i = 0; i < C3 / 32; ++i)
    sQ[w][lane + 32 * i] = (float)qr[lane + 32 * i];
  __syncthreads();
  const float scale = 0.1767766952966369f;          // 32^-0.5
  for (int e = lane; e < NHD * NHD; e += 32) {
    int i = e / NHD, j = e - (e / NHD) * NHD;
    const float* q = &sQ[w][i * HDD];
    const float* k = &sQ[w][CC + j * HDD];
    float s = 0.f;
#pragma unroll
    for (int d = 0; d < HDD; ++d) s += q[d] * k[d];
    sA[w][e] = s * scale;
  }
  __syncthreads();
  if (lane < NHD) {
    float mx = -1e30f;
#pragma unroll
    for (int j = 0; j < NHD; ++j) mx = fmaxf(mx, sA[w][lane * NHD + j]);
    float ev[NHD]; float sm = 0.f;
#pragma unroll
    for (int j = 0; j < NHD; ++j) { ev[j] = expf(sA[w][lane * NHD + j] - mx); sm += ev[j]; }
    float inv = 1.0f / sm;
#pragma unroll
    for (int j = 0; j < NHD; ++j) sA[w][lane * NHD + j] = ev[j] * inv;
  }
  __syncthreads();
  const int d = lane;                               // lane owns head-dim slot d
  float vals[NHD]; float ssum = 0.f, ssq = 0.f;
#pragma unroll
  for (int i = 0; i < NHD; ++i) {
    float o = 0.f;
#pragma unroll
    for (int j = 0; j < NHD; ++j) o += sA[w][i * NHD + j] * sQ[w][2 * CC + j * HDD + d];
    int c = d * NHD + i;                            // o.transpose(0,2,1) layout
    float nv = xt[n * CC + c] + gamma1[c] * o;
    vals[i] = nv;
    xt[n * CC + c] = nv;                            // residual stream updated in place
    ssum += nv; ssq += nv * nv;
  }
  ssum = wred(ssum); ssq = wred(ssq);
  float mu = ssum * (1.0f / CC);
  float rstd = rsqrtf(ssq * (1.0f / CC) - mu * mu + 1e-5f);
#pragma unroll
  for (int i = 0; i < NHD; ++i) {
    int c = d * NHD + i;
    y2[n * CC + c] = (half_t)((vals[i] - mu) * rstd * g2[c] + b2[c]);
  }
}

// ---------- WMMA GEMM: D(N x MM) = A(N x KK, f16) * W(MM x KK, f16)^T ----------
// Block = 8 waves, 128 rows x 64 cols. The 64xK B-panel is shared by all waves:
// staged in LDS (K-chunks of 384, padded rows -> conflict-free ds_load_b128).
// Inner loop: A register-pipelined from global (+prefetch), B from LDS, so the
// 4 WMMAs per k-step depend on DScnt (short) instead of full loadcnt drains.
template <int EPI, int KK, int MM>
__global__ void __launch_bounds__(256) k_gemm(
    const half_t* __restrict__ A, const half_t* __restrict__ Bw,
    const float* __restrict__ bias,
    const float* __restrict__ c0, const float* __restrict__ c1,
    const float* __restrict__ c2, const float* __restrict__ c3,
    const float* __restrict__ c4,
    const float* __restrict__ s0p, const float* __restrict__ s1p,
    const float* __restrict__ aux,
    float* __restrict__ of32, half_t* __restrict__ of16) {
  __shared__ half_t sB[64][KCH + KPAD];             // 50,176 B
  const int tid = threadIdx.x;
  const int wv = tid >> 5, lane = tid & 31;
  const long rbase = ((long)blockIdx.y * 8 + wv) * 16;
  const int cbase = blockIdx.x * 64;
  const int rsel = lane & 15;
  const int koff = (lane >> 4) << 3;                // K-chunk select per half-wave
  const half_t* Ar = A + (rbase + rsel) * (long)KK + koff;

  v8f acc[4] = {};
  for (int kc = 0; kc < KK; kc += KCH) {
    if (kc) __syncthreads();                        // protect previous chunk reads
    // cooperative B-panel fill: 64 rows x 384 halfs, b128 per thread, coalesced
    for (int e = tid; e < 64 * (KCH / 8); e += 256) {
      int row = e / (KCH / 8);
      int k8  = e - row * (KCH / 8);
      *(f4*)&sB[row][k8 * 8] =
          *(const f4*)(Bw + (long)(cbase + row) * KK + kc + k8 * 8);
    }
    __syncthreads();
    if (kc + KCH < KK) __builtin_prefetch(Ar + kc + KCH, 0, 1);   // next A chunk

    v16h a = ldtile(Ar + kc);
#pragma unroll
    for (int kk = 0; kk < KCH; kk += 32) {
      v16h an;
      if (kk + 32 < KCH) an = ldtile(Ar + kc + kk + 32);
      v16h b0 = ldtile(&sB[rsel     ][kk + koff]);
      v16h b1 = ldtile(&sB[rsel + 16][kk + koff]);
      v16h b2 = ldtile(&sB[rsel + 32][kk + koff]);
      v16h b3 = ldtile(&sB[rsel + 48][kk + koff]);
      acc[0] = __builtin_amdgcn_wmma_f32_16x16x32_f16(false, a, false, b0, (short)0, acc[0], false, false);
      acc[1] = __builtin_amdgcn_wmma_f32_16x16x32_f16(false, a, false, b1, (short)0, acc[1], false, false);
      acc[2] = __builtin_amdgcn_wmma_f32_16x16x32_f16(false, a, false, b2, (short)0, acc[2], false, false);
      acc[3] = __builtin_amdgcn_wmma_f32_16x16x32_f16(false, a, false, b3, (short)0, acc[3], false, false);
      if (kk + 32 < KCH) a = an;
    }
  }

  const int hi = koff;                              // row offset for lanes 16-31
  const float s0 = s0p ? s0p[0] : 0.f;
  const float s1 = s1p ? s1p[0] : 0.f;
#pragma unroll
  for (int t = 0; t < 4; ++t) {
    const int m = cbase + t * 16 + rsel;
#pragma unroll
    for (int vi = 0; vi < 8; ++vi) {
      const long r = rbase + hi + vi;
      float v = acc[t][vi];
      if (EPI == 0) {                               // qkv projection
        of16[r * (long)MM + m] = (half_t)(v + bias[m]);
      } else if (EPI == 1) {                        // embed -> u (f32) + fc1 input (f16)
        float t0 = v + bias[m];
        float u = t0 + c0[m] * (s0 * (c1[m] * t0 + c2[m]) + s1);
        of32[r * (long)CC + m] = u;
        of16[r * (long)CC + m] = (half_t)(c3[m] * u + c4[m]);
      } else if (EPI == 2) {                        // fc1 + exact GELU
        float h = v + bias[m];
        float ge = 0.5f * h * (1.0f + erff(h * 0.70710678118654752f));
        of16[r * (long)MM + m] = (half_t)ge;
      } else if (EPI == 3) {                        // fc2 -> w = affout_g*(u+ls_b*..)+affout_b
        float vv = aux[r * (long)CC + m] + c0[m] * (v + bias[m]);
        of16[r * (long)CC + m] = (half_t)(c1[m] * vv + c2[m]);
      } else {                                      // head -> residual + permuted store
        float mm2 = v + bias[m];
        float f = aux[r * (long)CC + m] + c0[m] * mm2;
        long bb = r / DHW;
        long s = r - bb * DHW;
        of32[(bb * CC + m) * (long)DHW + s] = f;
      }
    }
  }
}

extern "C" void kernel_launch(void* const* d_in, const int* in_sizes, int n_in,
                              void* d_out, int out_size, void* d_ws, size_t ws_size,
                              hipStream_t stream) {
  const float* x       = (const float*)d_in[0];
  const float* ln1_g   = (const float*)d_in[1];
  const float* ln1_b   = (const float*)d_in[2];
  const float* ln2_g   = (const float*)d_in[3];
  const float* ln2_b   = (const float*)d_in[4];
  const float* w_qkv   = (const float*)d_in[5];
  const float* b_qkv   = (const float*)d_in[6];
  const float* gamma1  = (const float*)d_in[7];
  const float* gamma2  = (const float*)d_in[8];
  const float* embed_w = (const float*)d_in[9];
  const float* embed_b = (const float*)d_in[10];
  const float* aff1_g  = (const float*)d_in[11];
  const float* aff1_b  = (const float*)d_in[12];
  const float* tok_w   = (const float*)d_in[13];
  const float* tok_b   = (const float*)d_in[14];
  const float* ls_a    = (const float*)d_in[15];
  const float* aff2_g  = (const float*)d_in[16];
  const float* aff2_b  = (const float*)d_in[17];
  const float* fc1_w   = (const float*)d_in[18];
  const float* fc1_b   = (const float*)d_in[19];
  const float* fc2_w   = (const float*)d_in[20];
  const float* fc2_b   = (const float*)d_in[21];
  const float* ls_b    = (const float*)d_in[22];
  const float* affout_g= (const float*)d_in[23];
  const float* affout_b= (const float*)d_in[24];
  const float* head_w  = (const float*)d_in[25];
  const float* head_b  = (const float*)d_in[26];
  (void)in_sizes; (void)n_in; (void)out_size; (void)ws_size;

  char* ws = (char*)d_ws;
  size_t off = 0;
  auto alloc = [&](size_t bytes) -> void* {
    void* p = ws + off;
    off += (bytes + 255) & ~(size_t)255;
    return p;
  };
  half_t* wqkv16  = (half_t*)alloc((size_t)C3 * CC * 2);
  half_t* wemb16  = (half_t*)alloc((size_t)CC * CC * 2);
  half_t* wfc1_16 = (half_t*)alloc((size_t)HID * CC * 2);
  half_t* wfc2_16 = (half_t*)alloc((size_t)CC * HID * 2);
  half_t* whead16 = (half_t*)alloc((size_t)CC * CC * 2);
  float*  xt      = (float*) alloc((size_t)NTOK * CC * 4);   // residual stream
  half_t* y16     = (half_t*)alloc((size_t)NTOK * CC * 2);   // LN out / w16 (reused)
  half_t* qkv16   = (half_t*)alloc((size_t)NTOK * C3 * 2);
  float*  ubuf    = (float*) alloc((size_t)NTOK * CC * 4);
  half_t* a16     = (half_t*)alloc((size_t)NTOK * CC * 2);   // fc1 input
  half_t* h16     = (half_t*)alloc((size_t)NTOK * HID * 2);  // fc2 input

  auto cvt = [&](const float* s, half_t* dd, int n) {
    k_cvt<<<(n + 255) / 256, 256, 0, stream>>>(s, dd, n);
  };
  cvt(w_qkv,   wqkv16,  C3 * CC);
  cvt(embed_w, wemb16,  CC * CC);
  cvt(fc1_w,   wfc1_16, HID * CC);
  cvt(fc2_w,   wfc2_16, CC * HID);
  cvt(head_w,  whead16, CC * CC);

  k_ln1<<<(int)(NTOK / 32), 256, 0, stream>>>(x, xt, y16, ln1_g, ln1_b);

  dim3 blk(256);
  dim3 gq(C3 / 64, (int)(NTOK / 128));
  k_gemm<0, CC, C3><<<gq, blk, 0, stream>>>(y16, wqkv16, b_qkv,
      nullptr, nullptr, nullptr, nullptr, nullptr, nullptr, nullptr, nullptr,
      nullptr, qkv16);

  k_attn<<<(int)(NTOK / 4), 128, 0, stream>>>(qkv16, xt, gamma1, ln2_g, ln2_b, y16);

  dim3 gc(CC / 64, (int)(NTOK / 128));
  k_gemm<1, CC, CC><<<gc, blk, 0, stream>>>(y16, wemb16, embed_b,
      ls_a, aff1_g, aff1_b, aff2_g, aff2_b, tok_w, tok_b, nullptr,
      ubuf, a16);

  dim3 gh(HID / 64, (int)(NTOK / 128));
  k_gemm<2, CC, HID><<<gh, blk, 0, stream>>>(a16, wfc1_16, fc1_b,
      nullptr, nullptr, nullptr, nullptr, nullptr, nullptr, nullptr, nullptr,
      nullptr, h16);

  k_gemm<3, HID, CC><<<gc, blk, 0, stream>>>(h16, wfc2_16, fc2_b,
      ls_b, affout_g, affout_b, nullptr, nullptr, nullptr, nullptr, ubuf,
      nullptr, y16);

  k_gemm<4, CC, CC><<<gc, blk, 0, stream>>>(y16, whead16, head_b,
      gamma2, nullptr, nullptr, nullptr, nullptr, nullptr, nullptr, xt,
      (float*)d_out, nullptr);
}
